// EmbedCoeffs_4140348473500
// MI455X (gfx1250) — matrix operand order repaired and checked by reading, not compiled, not run on hardware
//
#include <hip/hip_runtime.h>

typedef __attribute__((ext_vector_type(16))) _Float16 v16h;
typedef __attribute__((ext_vector_type(8)))  float    v8f;

#define N_NODE 50000
#define N_EDGE 800000
#define N_ORB  4
#define Z_DIM  128
#define E_DIM  64
#define HID    128

__device__ __forceinline__ float silu_f(float x) {
    return x * (1.0f / (1.0f + __expf(-x)));
}

__device__ __forceinline__ v8f wmma_f16(v16h a, v16h b, v8f c) {
    // D = A(16x32 f16) * B(32x16 f16) + C(16x16 f32)
    return __builtin_amdgcn_wmma_f32_16x16x32_f16(
        /*neg_a=*/false, a, /*neg_b=*/false, b,
        /*c_mod=*/(short)0, c, /*reuse_a=*/false, /*reuse_b=*/false);
}

// Load A fragment (16x32, f16) from an LDS tile stored row-major with `pitch` halves.
// ISA layout: lane L holds row M=L&15; halves 0-7 = K (L>=16?8:0)+0..7, halves 8-15 = +16.
__device__ __forceinline__ v16h load_afrag(const _Float16* As, int pitch,
                                           int mbase, int kbase, int lane) {
    int row = mbase + (lane & 15);
    int k0  = kbase + ((lane >> 4) << 3);
    const _Float16* p0 = As + row * pitch + k0;       // 16B aligned (pitch%8==0, k0%8==0)
    const _Float16* p1 = p0 + 16;
    v16h a;
#pragma unroll
    for (int i = 0; i < 8; ++i) { a[i] = p0[i]; a[8 + i] = p1[i]; }
    return a;
}

// ---------------------------------------------------------------------------
// Kernel 0: pack W (row-major [128][K]) transposed into WMMA B-fragment order:
//   out[((kt*8 + nt)*32 + lane)*16 + i] = W[(16*nt + (lane&15))*K + 32*kt + (lane>>4)*8 + (i&7) + (i>>3)*16]
// so a wave reads its 32x16 B tile as one contiguous v16h (32B) per lane.
// ---------------------------------------------------------------------------
__device__ __forceinline__ void pack_one(const float* __restrict__ W,
                                         _Float16* __restrict__ out,
                                         int K, int tid, int nthreads) {
    int total = K * 128;
    for (int t = tid; t < total; t += nthreads) {
        int i    = t & 15;
        int lane = (t >> 4) & 31;
        int nt   = (t >> 9) & 7;
        int kt   = t >> 12;
        int n = 16 * nt + (lane & 15);
        int k = 32 * kt + ((lane >> 4) << 3) + (i & 7) + ((i >> 3) << 4);
        out[t] = (_Float16)W[n * K + k];
    }
}

__global__ __launch_bounds__(256)
void pack_weights_kernel(const float* __restrict__ zW1, const float* __restrict__ zW2,
                         const float* __restrict__ eW1, const float* __restrict__ eW2,
                         _Float16* __restrict__ zW1p, _Float16* __restrict__ zW2p,
                         _Float16* __restrict__ eW1p, _Float16* __restrict__ eW2p) {
    int tid = blockIdx.x * blockDim.x + threadIdx.x;
    int nth = gridDim.x * blockDim.x;
    pack_one(zW1, zW1p, 2 * Z_DIM, tid, nth);
    pack_one(zW2, zW2p, HID,      tid, nth);
    pack_one(eW1, eW1p, E_DIM,    tid, nth);
    pack_one(eW2, eW2p, HID,      tid, nth);
}

// ---------------------------------------------------------------------------
// Kernel 1: per-node e-branch MLP.
// Rows = node*4 + orb (200000 rows). 64 rows per block, 8 waves.
//   e2[row] = silu(silu(e_embed[row]) @ eW1^T) @ eW2^T
// ---------------------------------------------------------------------------
__global__ __launch_bounds__(256)
void e_mlp_kernel(const float* __restrict__ e_embed,
                  const _Float16* __restrict__ eW1p,
                  const _Float16* __restrict__ eW2p,
                  float* __restrict__ e2) {
    const int P1 = E_DIM + 8;   // 72 halves/row  (144B, 16B aligned)
    const int P2 = HID + 8;     // 136 halves/row (272B, 16B aligned)
    __shared__ __align__(16) _Float16 smem[64 * 136];   // 17408B, reused A1 -> A2

    int tid  = threadIdx.x;
    int lane = tid & 31;
    int wave = tid >> 5;
    long long row0 = (long long)blockIdx.x * 64;

    // stage A1 = f16(silu(e_embed tile))  (64 x 64)
    {
        const float* src = e_embed + row0 * E_DIM;
        for (int t = tid; t < 64 * E_DIM; t += 256) {
            int r = t >> 6, c = t & 63;
            smem[r * P1 + c] = (_Float16)silu_f(src[r * E_DIM + c]);
        }
    }
    __syncthreads();

    int nt   = wave;                       // this wave's 16-wide N tile
    int colN = nt * 16 + (lane & 15);
    int mro  = (lane >> 4) << 3;           // 0 or 8: D-matrix row offset per lane half

    v8f zero = {0.f, 0.f, 0.f, 0.f, 0.f, 0.f, 0.f, 0.f};
    v8f acc[4] = {zero, zero, zero, zero};

    // GEMM1: (64x64) @ (64x128), K loop of 2
#pragma unroll
    for (int kt = 0; kt < 2; ++kt) {
        v16h b = *(const v16h*)(eW1p + (((kt * 8 + nt) * 32 + lane) << 4));
#pragma unroll
        for (int mt = 0; mt < 4; ++mt) {
            v16h a = load_afrag(smem, P1, mt * 16, kt * 32, lane);
            acc[mt] = wmma_f16(a, b, acc[mt]);
        }
    }
    __syncthreads();

    // silu -> A2 (64 x 128 f16)
#pragma unroll
    for (int mt = 0; mt < 4; ++mt)
#pragma unroll
        for (int r = 0; r < 8; ++r) {
            int row = mt * 16 + mro + r;
            smem[row * P2 + colN] = (_Float16)silu_f(acc[mt][r]);
        }
    __syncthreads();

    // GEMM2: (64x128) @ (128x128), K loop of 4
#pragma unroll
    for (int mt = 0; mt < 4; ++mt) acc[mt] = zero;
#pragma unroll
    for (int kt = 0; kt < 4; ++kt) {
        v16h b = *(const v16h*)(eW2p + (((kt * 8 + nt) * 32 + lane) << 4));
#pragma unroll
        for (int mt = 0; mt < 4; ++mt) {
            v16h a = load_afrag(smem, P2, mt * 16, kt * 32, lane);
            acc[mt] = wmma_f16(a, b, acc[mt]);
        }
    }

    // store f32 result tile to workspace (stays L2-resident, 102MB < 192MB L2)
#pragma unroll
    for (int mt = 0; mt < 4; ++mt)
#pragma unroll
        for (int r = 0; r < 8; ++r) {
            long long row = row0 + mt * 16 + mro + r;
            e2[row * HID + colN] = acc[mt][r];
        }
}

// ---------------------------------------------------------------------------
// Kernel 2: z-branch (two dense layers over gathered+concat endpoints) fused
// with the final combine: out[e,orb,:] = e2[idx_j[e],orb,:] * (1 + z[e,:])
// 64 edges per block, 8 waves.
// ---------------------------------------------------------------------------
__global__ __launch_bounds__(256)
void z_combine_kernel(const float* __restrict__ z_embed,
                      const int* __restrict__ idx_i,
                      const int* __restrict__ idx_j,
                      const _Float16* __restrict__ zW1p,
                      const float* __restrict__ zb1,
                      const _Float16* __restrict__ zW2p,
                      const float* __restrict__ zb2,
                      const float* __restrict__ e2,
                      float* __restrict__ out) {
    const int P1 = 2 * Z_DIM + 8;   // 264 halves/row (528B, 16B aligned)
    const int P2 = HID + 8;         // 136 halves/row
    // 33792B buffer reused: A1 (64x264 f16) -> A2 (64x136 f16) -> Zscale (64x128 f32)
    __shared__ __align__(16) unsigned char smem_raw[64 * 264 * 2];
    _Float16* As = (_Float16*)smem_raw;
    float*    Zs = (float*)smem_raw;
    __shared__ int s_ii[64];
    __shared__ int s_jj[64];

    int tid  = threadIdx.x;
    int lane = tid & 31;
    int wave = tid >> 5;
    int e0   = blockIdx.x * 64;

    if (tid < 64) { s_ii[tid] = idx_i[e0 + tid]; s_jj[tid] = idx_j[e0 + tid]; }
    __syncthreads();

    // gather both endpoints, silu, f16 -> A1 (64 x 256); z_embed is L2-resident
    for (int t = tid; t < 64 * 256; t += 256) {
        int r = t >> 8, c = t & 255;
        int node = (c < 128) ? s_ii[r] : s_jj[r];
        float v = z_embed[(long long)node * Z_DIM + (c & 127)];
        As[r * P1 + c] = (_Float16)silu_f(v);
    }
    __syncthreads();

    int nt   = wave;
    int colN = nt * 16 + (lane & 15);
    int mro  = (lane >> 4) << 3;
    float b1v = zb1[colN];
    float b2v = zb2[colN];

    v8f zero = {0.f, 0.f, 0.f, 0.f, 0.f, 0.f, 0.f, 0.f};
    v8f acc[4] = {zero, zero, zero, zero};

    // GEMM1: (64x256) @ (256x128), K loop of 8
#pragma unroll
    for (int kt = 0; kt < 8; ++kt) {
        v16h b = *(const v16h*)(zW1p + (((kt * 8 + nt) * 32 + lane) << 4));
#pragma unroll
        for (int mt = 0; mt < 4; ++mt) {
            v16h a = load_afrag(As, P1, mt * 16, kt * 32, lane);
            acc[mt] = wmma_f16(a, b, acc[mt]);
        }
    }
    __syncthreads();

    // h = silu(c + zb1) -> A2
#pragma unroll
    for (int mt = 0; mt < 4; ++mt)
#pragma unroll
        for (int r = 0; r < 8; ++r) {
            int row = mt * 16 + mro + r;
            As[row * P2 + colN] = (_Float16)silu_f(acc[mt][r] + b1v);
        }
    __syncthreads();

    // GEMM2: (64x128) @ (128x128), K loop of 4
#pragma unroll
    for (int mt = 0; mt < 4; ++mt) acc[mt] = zero;
#pragma unroll
    for (int kt = 0; kt < 4; ++kt) {
        v16h b = *(const v16h*)(zW2p + (((kt * 8 + nt) * 32 + lane) << 4));
#pragma unroll
        for (int mt = 0; mt < 4; ++mt) {
            v16h a = load_afrag(As, P2, mt * 16, kt * 32, lane);
            acc[mt] = wmma_f16(a, b, acc[mt]);
        }
    }
    __syncthreads();

    // scale = 1 + z  -> LDS (f32)
#pragma unroll
    for (int mt = 0; mt < 4; ++mt)
#pragma unroll
        for (int r = 0; r < 8; ++r) {
            int row = mt * 16 + mro + r;
            Zs[row * HID + colN] = 1.0f + acc[mt][r] + b2v;
        }
    __syncthreads();

    // fused combine: coalesced stores of the 1.64GB output (the bandwidth roof)
    for (int t = tid; t < 64 * N_ORB * HID; t += 256) {
        int r   = t >> 9;
        int rem = t & 511;
        int orb = rem >> 7;
        int c   = rem & 127;
        long long erow = (long long)s_jj[r] * N_ORB + orb;
        float ev = e2[erow * HID + c];                 // L2 hit
        float sc = Zs[r * HID + c];
        out[((long long)(e0 + r) * N_ORB + orb) * HID + c] = ev * sc;
    }
}

// ---------------------------------------------------------------------------
extern "C" void kernel_launch(void* const* d_in, const int* in_sizes, int n_in,
                              void* d_out, int out_size, void* d_ws, size_t ws_size,
                              hipStream_t stream) {
    const float* z_embed = (const float*)d_in[0];
    const float* e_embed = (const float*)d_in[1];
    const int*   idx_i   = (const int*)d_in[2];
    const int*   idx_j   = (const int*)d_in[3];
    const float* zW1     = (const float*)d_in[4];
    const float* zb1     = (const float*)d_in[5];
    const float* zW2     = (const float*)d_in[6];
    const float* zb2     = (const float*)d_in[7];
    const float* eW1     = (const float*)d_in[8];
    const float* eW2     = (const float*)d_in[9];
    float* out = (float*)d_out;

    // workspace layout (all offsets 256B-aligned)
    unsigned char* ws = (unsigned char*)d_ws;
    size_t off = 0;
    float*    e2   = (float*)(ws + off);    off += (size_t)N_NODE * N_ORB * HID * sizeof(float); // 102.4 MB
    _Float16* zW1p = (_Float16*)(ws + off); off += (size_t)(2 * Z_DIM) * HID * sizeof(_Float16);
    _Float16* zW2p = (_Float16*)(ws + off); off += (size_t)HID * HID * sizeof(_Float16);
    _Float16* eW1p = (_Float16*)(ws + off); off += (size_t)E_DIM * HID * sizeof(_Float16);
    _Float16* eW2p = (_Float16*)(ws + off); off += (size_t)HID * HID * sizeof(_Float16);

    pack_weights_kernel<<<32, 256, 0, stream>>>(zW1, zW2, eW1, eW2,
                                                zW1p, zW2p, eW1p, eW2p);
    e_mlp_kernel<<<(N_NODE * N_ORB) / 64, 256, 0, stream>>>(e_embed, eW1p, eW2p, e2);
    z_combine_kernel<<<N_EDGE / 64, 256, 0, stream>>>(z_embed, idx_i, idx_j,
                                                      zW1p, zb1, zW2p, zb2, e2, out);
}
